// MultiheadCosformerAttention_63419487093380
// MI455X (gfx1250) — matrix operand order repaired and checked by reading, not compiled, not run on hardware
//
#include <hip/hip_runtime.h>
#include <hip/hip_bf16.h>
#include <math.h>

// ---------------------------------------------------------------------------
// cosformer attention, CDNA5 (gfx1250, wave32).
// All GEMMs on v_wmma_f32_16x16x32_bf16.  cos/sin positional reweighting is
// folded into the projection epilogue (prescaled bf16 buffers), so the stage
// B/C hot loops are pure ds_load_b128 -> v_wmma with async global->LDS staging.
// L=2048, B=4, E=1024, H=16, HD=64.
// ---------------------------------------------------------------------------

typedef __attribute__((ext_vector_type(16))) __bf16 v16bf;
typedef __attribute__((ext_vector_type(8)))  __bf16 v8bf;
typedef __attribute__((ext_vector_type(4)))  __bf16 v4bf;
typedef __attribute__((ext_vector_type(8)))  float  v8f;

#define L_SEQ   2048
#define B_SZ    4
#define E_DIM   1024
#define H_NUM   16
#define HD_DIM  64
#define M_TOT   (L_SEQ * B_SZ)      // 8192
#define NH      (B_SZ * H_NUM)      // 64 heads
#define BH_STRIDE (B_SZ * E_DIM)    // 4096 elements per l slice

__constant__ const float ANG = 1.5707963267948966f / 2048.0f;  // (pi/2)/L

__device__ __forceinline__ int head_offset(int n) {
    return (n >> 4) * E_DIM + (n & 15) * HD_DIM;   // (n/16)*1024 + (n%16)*64
}

// ---- CDNA5 async global->LDS (ASYNCcnt path) -------------------------------
__device__ __forceinline__ void async_load_b128(void* lds_dst, const void* gsrc) {
    unsigned lds_off = (unsigned)(unsigned long long)lds_dst;  // low 32 = LDS offset
    asm volatile("global_load_async_to_lds_b128 %0, %1, off"
                 :: "v"(lds_off), "v"(gsrc) : "memory");
}
__device__ __forceinline__ void wait_async0() {
    asm volatile("s_wait_asynccnt 0x0" ::: "memory");
}

// ---- fragment builders -----------------------------------------------------
// A 16x32 bf16 layout: lanes 0-15 K={0..7,16..23}, lanes 16-31 K={8..15,24..31}
__device__ __forceinline__ v16bf frag_a32(const __bf16* seg, int hi) {
    const v8bf* p = (const v8bf*)seg;
    v8bf lo = p[hi];
    v8bf hh = p[2 + hi];
    return __builtin_shufflevector(lo, hh, 0,1,2,3,4,5,6,7,8,9,10,11,12,13,14,15);
}
// B 32x16 bf16 layout: lane%16 = N column, lanes 0-15 K={0..7,16..23}, lanes 16-31 K={8..15,24..31} (row stride 64).
__device__ __forceinline__ v16bf frag_b16(const __bf16* seg, int hi) {
    v16bf b;
#pragma unroll
    for (int s = 0; s < 16; ++s) {
        int k = (s < 8) ? (8 * hi + s) : (16 + 8 * hi + (s - 8));
        b[s] = seg[k * 64];
    }
    return b;
}

__device__ __forceinline__ v8f wmma_bf16(const v16bf& a, const v16bf& b, const v8f& c) {
    return __builtin_amdgcn_wmma_f32_16x16x32_bf16(false, a, false, b, (short)0, c,
                                                   false, false);
}

// ---------------------------------------------------------------------------
// Stage A: proj = X @ W^T + bias.
//   mode 0: Yc = proj (bf16)                                  [for V]
//   mode 1: Yc = relu(proj)*cos(l), Ys = relu(proj)*sin(l)    [for Q, K]
// 256 thr (8 waves), 128x64 tile, K-step 32, double-buffered bf16 LDS.
// Each wave: 1 B-fragment shared by 4 M-subtiles -> 4 WMMA per K-step.
// Positional weights via a 32-entry LDS trig table (one sincosf per lane).
// ---------------------------------------------------------------------------
__global__ __launch_bounds__(256) void cosformer_proj_gemm(
    const float* __restrict__ X, const float* __restrict__ W,
    const float* __restrict__ bias, __bf16* __restrict__ Yc,
    __bf16* __restrict__ Ys, int mode)
{
    __shared__ __bf16 As[2][128][32];  // [m][k] natural-k-order rows (8KB each)
    __shared__ __bf16 Bs[2][32][64];   // [k][n] (4KB each)
    __shared__ float cwT[32], swT[32]; // per-l trig table (128 rows = 32 l's)

    const int tid = threadIdx.x;
    const int wave = tid >> 5;
    const int lane = tid & 31;
    const int lane_lo = lane & 15;
    const int hi = lane >> 4;
    const int tn  = wave & 3;          // N sub-tile
    const int tmb = wave >> 2;         // 0..1; M sub-tiles tmb, tmb+2, +4, +6

    const int m0 = blockIdx.x * 128;
    const int n0 = blockIdx.y * 64;

    const int rx = tid >> 3;           // 0..31
    const int cc = (tid & 7) * 4;      // 0..28

    v8f acc[4] = {{}, {}, {}, {}};
    float4 xa[4], wb[2];

    auto fetch = [&](int k0) {
#pragma unroll
        for (int j = 0; j < 4; ++j)
            xa[j] = *(const float4*)(X + (m0 + rx + 32 * j) * E_DIM + k0 + cc);
#pragma unroll
        for (int j = 0; j < 2; ++j)
            wb[j] = *(const float4*)(W + (n0 + rx + 32 * j) * E_DIM + k0 + cc);
    };
    auto stash = [&](int buf) {
#pragma unroll
        for (int j = 0; j < 4; ++j)
            *(v4bf*)&As[buf][rx + 32 * j][cc] =
                (v4bf){(__bf16)xa[j].x, (__bf16)xa[j].y,
                       (__bf16)xa[j].z, (__bf16)xa[j].w};
#pragma unroll
        for (int j = 0; j < 2; ++j) {
            Bs[buf][cc + 0][rx + 32 * j] = (__bf16)wb[j].x;
            Bs[buf][cc + 1][rx + 32 * j] = (__bf16)wb[j].y;
            Bs[buf][cc + 2][rx + 32 * j] = (__bf16)wb[j].z;
            Bs[buf][cc + 3][rx + 32 * j] = (__bf16)wb[j].w;
        }
    };

    fetch(0);
    stash(0);
    __syncthreads();

    int buf = 0;
    for (int k0 = 0; k0 < E_DIM; k0 += 32) {
        const int nk = k0 + 32;
        if (nk < E_DIM) fetch(nk);                  // overlaps WMMA below

        v16bf bfrag = frag_b16(&Bs[buf][0][tn * 16 + lane_lo], hi);
#pragma unroll
        for (int j = 0; j < 4; ++j) {
            v16bf a = frag_a32(&As[buf][(tmb + 2 * j) * 16 + lane_lo][0], hi);
            acc[j] = wmma_bf16(a, bfrag, acc[j]);
        }

        if (nk < E_DIM) stash(buf ^ 1);
        __syncthreads();
        buf ^= 1;
    }

    // one sincosf per lane -> 32-entry table covering this block's l range
    if (tid < 32) {
        float sw, cw;
        sincosf((float)((m0 >> 2) + tid + 1) * ANG, &sw, &cw);
        cwT[tid] = cw;
        swT[tid] = sw;
    }
    __syncthreads();

    const int col = n0 + tn * 16 + lane_lo;
    const float bv = bias[col];
#pragma unroll
    for (int j = 0; j < 4; ++j) {
#pragma unroll
        for (int i = 0; i < 8; ++i) {
            int rloc = (tmb + 2 * j) * 16 + hi * 8 + i;       // 0..127
            int off  = (m0 + rloc) * E_DIM + col;             // < 2^23, int ok
            float v = acc[j][i] + bv;
            if (mode == 0) {
                Yc[off] = (__bf16)v;
            } else {
                v = fmaxf(v, 0.0f);
                int lloc = rloc >> 2;                         // 0..31
                Yc[off] = (__bf16)(v * cwT[lloc]);
                Ys[off] = (__bf16)(v * swT[lloc]);
            }
        }
    }
}

// ---------------------------------------------------------------------------
// Stage B: per head n:  kvT_cos[d][m] = sum_s Kc[n,s,d]*V[n,s,m]  (and sin)
// A = V^T (64 x 2048), B = Kc / Ks (2048 x 64), both prescaled bf16.
// Kc/Ks tiles staged with async b128 (layout-natural); V register-staged +
// transposed store.  Double-buffered; hot loop is pure ds_load -> wmma.
// ---------------------------------------------------------------------------
__global__ __launch_bounds__(256) void cosformer_kv_gemm(
    const __bf16* __restrict__ Kc, const __bf16* __restrict__ Ksn,
    const __bf16* __restrict__ Vp, __bf16* __restrict__ kvcT,
    __bf16* __restrict__ kvsT)
{
    __shared__ __bf16 KcS[2][32][64];  // [s][d]  (async dest)
    __shared__ __bf16 KsS[2][32][64];  // [s][d]  (async dest)
    __shared__ __bf16 Vt[2][64][32];   // [m][s]  (transposed)

    const int n = blockIdx.x;
    const int hoff = head_offset(n);

    const int tid = threadIdx.x;
    const int wave = tid >> 5;
    const int lane = tid & 31;
    const int lane_lo = lane & 15;
    const int hi = lane >> 4;
    const int tn  = wave & 3;          // d sub-tile
    const int tm0 = wave >> 2;         // m sub-tiles
    const int tm1 = tm0 + 2;

    const int r  = tid >> 3;           // 0..31 (s row)
    const int ch = tid & 7;            // 16B chunk

    v8f cc0 = {}, cc1 = {}, cs0 = {}, cs1 = {};
    v8bf vreg;

    auto issue_k = [&](int s0, int buf) {
        int off = (s0 + r) * BH_STRIDE + hoff + ch * 8;       // int offsets
        async_load_b128(&KcS[buf][r][ch * 8], Kc + off);
        async_load_b128(&KsS[buf][r][ch * 8], Ksn + off);
    };
    auto fetch_v = [&](int s0) {
        vreg = *(const v8bf*)(Vp + (s0 + r) * BH_STRIDE + hoff + ch * 8);
    };
    auto stash_v = [&](int buf) {
#pragma unroll
        for (int j = 0; j < 8; ++j) Vt[buf][ch * 8 + j][r] = vreg[j];
    };

    issue_k(0, 0);
    fetch_v(0);
    stash_v(0);
    wait_async0();
    __syncthreads();

    int buf = 0;
    for (int s0 = 0; s0 < L_SEQ; s0 += 32) {
        const int ns = s0 + 32;
        if (ns < L_SEQ) { issue_k(ns, buf ^ 1); fetch_v(ns); }

        v16bf bcos = frag_b16(&KcS[buf][0][tn * 16 + lane_lo], hi);
        v16bf bsin = frag_b16(&KsS[buf][0][tn * 16 + lane_lo], hi);
        v16bf a0 = frag_a32(&Vt[buf][tm0 * 16 + lane_lo][0], hi);
        v16bf a1 = frag_a32(&Vt[buf][tm1 * 16 + lane_lo][0], hi);
        cc0 = wmma_bf16(a0, bcos, cc0);
        cs0 = wmma_bf16(a0, bsin, cs0);
        cc1 = wmma_bf16(a1, bcos, cc1);
        cs1 = wmma_bf16(a1, bsin, cs1);

        if (ns < L_SEQ) stash_v(buf ^ 1);
        wait_async0();
        __syncthreads();
        buf ^= 1;
    }

    // store transposed [d][m] so stage C B-fragments are plain row loads
    const int d = tn * 16 + lane_lo;
    const int base = n * (HD_DIM * HD_DIM);
#pragma unroll
    for (int i = 0; i < 8; ++i) {
        int mr0 = tm0 * 16 + hi * 8 + i;
        int mr1 = tm1 * 16 + hi * 8 + i;
        kvcT[base + d * HD_DIM + mr0] = (__bf16)cc0[i];
        kvcT[base + d * HD_DIM + mr1] = (__bf16)cc1[i];
        kvsT[base + d * HD_DIM + mr0] = (__bf16)cs0[i];
        kvsT[base + d * HD_DIM + mr1] = (__bf16)cs1[i];
    }
}

// ---------------------------------------------------------------------------
// Stage B2: ksum_cos[n][d] = sum_s Kc[n,s,d] (and sin).  One block per head.
// ---------------------------------------------------------------------------
__global__ __launch_bounds__(256) void cosformer_ksum(
    const __bf16* __restrict__ Kc, const __bf16* __restrict__ Ksn,
    float* __restrict__ sumcos, float* __restrict__ sumsin)
{
    __shared__ float pc[4][64];
    __shared__ float ps[4][64];

    const int n = blockIdx.x;
    const int hoff = head_offset(n);
    const int tid = threadIdx.x;
    const int d = tid & 63;
    const int part = tid >> 6;

    float ac = 0.0f, asn = 0.0f;
    for (int s = part; s < L_SEQ; s += 4) {
        int off = s * BH_STRIDE + hoff + d;
        ac  += (float)Kc[off];
        asn += (float)Ksn[off];
    }
    pc[part][d] = ac;
    ps[part][d] = asn;
    __syncthreads();
    if (tid < 64) {
        sumcos[n * 64 + tid] = pc[0][tid] + pc[1][tid] + pc[2][tid] + pc[3][tid];
        sumsin[n * 64 + tid] = ps[0][tid] + ps[1][tid] + ps[2][tid] + ps[3][tid];
    }
}

// ---------------------------------------------------------------------------
// Stage C: per head n, 64 l-rows per block:
//   z[l]     = 1/(sum_d Qc[l,d]*ksumc[d] + Qs[l,d]*ksums[d] + eps)
//   out[l,m] = z[l]*sum_d (Qc[l,d]*kvcT[d,m] + Qs[l,d]*kvsT[d,m])
// All four tiles staged with async b128 (prescaled bf16, no transform).
// ---------------------------------------------------------------------------
__global__ __launch_bounds__(256) void cosformer_attn_gemm(
    const __bf16* __restrict__ Qc, const __bf16* __restrict__ Qsn,
    const __bf16* __restrict__ kvcT, const __bf16* __restrict__ kvsT,
    const float* __restrict__ sumcos, const float* __restrict__ sumsin,
    float* __restrict__ out)
{
    __shared__ __bf16 QcS[64][64];     // [l][d]
    __shared__ __bf16 QsS[64][64];     // [l][d]
    __shared__ __bf16 KVc[64][64];     // [d][m]
    __shared__ __bf16 KVs[64][64];
    __shared__ float scos[64], ssin[64], zbuf[64];

    const int n  = blockIdx.y;
    const int l0 = blockIdx.x * 64;
    const int hoff = head_offset(n);

    const int tid = threadIdx.x;
    const int wave = tid >> 5;
    const int lane = tid & 31;
    const int lane_lo = lane & 15;
    const int hi = lane >> 4;
    const int tn  = wave & 3;          // m sub-tile
    const int tm0 = wave >> 2;         // l sub-tiles
    const int tm1 = tm0 + 2;

    const int kvbase = n * (HD_DIM * HD_DIM);
#pragma unroll
    for (int j = 0; j < 2; ++j) {
        int c  = tid + j * 256;        // 0..511
        int r  = c >> 3;               // 0..63
        int ch = (c & 7) * 8;          // bf16 offset of 16B chunk
        int qoff = (l0 + r) * BH_STRIDE + hoff + ch;
        async_load_b128(&QcS[r][ch], Qc  + qoff);
        async_load_b128(&QsS[r][ch], Qsn + qoff);
        async_load_b128(&KVc[r][ch], kvcT + kvbase + r * HD_DIM + ch);
        async_load_b128(&KVs[r][ch], kvsT + kvbase + r * HD_DIM + ch);
    }
    if (tid < 64) {
        scos[tid] = sumcos[n * 64 + tid];
        ssin[tid] = sumsin[n * 64 + tid];
    }
    wait_async0();
    __syncthreads();

    if (tid < 64) {
        float acc = 0.0f;
#pragma unroll
        for (int d = 0; d < 64; ++d)
            acc += (float)QcS[tid][d] * scos[d] + (float)QsS[tid][d] * ssin[d];
        zbuf[tid] = 1.0f / (acc + 1e-6f);
    }
    __syncthreads();

    v8f c0 = {}, c1 = {};
#pragma unroll
    for (int kstep = 0; kstep < 2; ++kstep) {
        const int k0 = kstep * 32;
        v16bf bcos = frag_b16(&KVc[k0][tn * 16 + lane_lo], hi);
        v16bf bsin = frag_b16(&KVs[k0][tn * 16 + lane_lo], hi);
        v16bf ac0 = frag_a32(&QcS[tm0 * 16 + lane_lo][k0], hi);
        v16bf as0 = frag_a32(&QsS[tm0 * 16 + lane_lo][k0], hi);
        v16bf ac1 = frag_a32(&QcS[tm1 * 16 + lane_lo][k0], hi);
        v16bf as1 = frag_a32(&QsS[tm1 * 16 + lane_lo][k0], hi);
        c0 = wmma_bf16(ac0, bcos, c0);
        c0 = wmma_bf16(as0, bsin, c0);
        c1 = wmma_bf16(ac1, bcos, c1);
        c1 = wmma_bf16(as1, bsin, c1);
    }

    const int m = tn * 16 + lane_lo;
    const int obase = n * (L_SEQ * HD_DIM);
#pragma unroll
    for (int i = 0; i < 8; ++i) {
        int lr0 = tm0 * 16 + hi * 8 + i;
        int lr1 = tm1 * 16 + hi * 8 + i;
        out[obase + (l0 + lr0) * HD_DIM + m] = c0[i] * zbuf[lr0];
        out[obase + (l0 + lr1) * HD_DIM + m] = c1[i] * zbuf[lr1];
    }
}

// ---------------------------------------------------------------------------
// Host launch.  Inputs: 0=query 1=key 2=value 3=Wq 4=bq 5=Wk 6=bk (fp32).
// Workspace: Qc,Qs,Kc,Ks,V bf16 (16MB each) + kvT bf16 (0.5MB x2) + sums.
// ---------------------------------------------------------------------------
extern "C" void kernel_launch(void* const* d_in, const int* in_sizes, int n_in,
                              void* d_out, int out_size, void* d_ws, size_t ws_size,
                              hipStream_t stream) {
    (void)in_sizes; (void)n_in; (void)out_size; (void)ws_size;
    const float* query = (const float*)d_in[0];
    const float* key   = (const float*)d_in[1];
    const float* value = (const float*)d_in[2];
    const float* Wq    = (const float*)d_in[3];
    const float* bq    = (const float*)d_in[4];
    const float* Wk    = (const float*)d_in[5];
    const float* bk    = (const float*)d_in[6];
    float* out = (float*)d_out;

    const size_t PROJ = (size_t)M_TOT * E_DIM;        // 8388608 elements
    const size_t KVSZ = (size_t)NH * HD_DIM * HD_DIM; // 262144 elements
    __bf16* Qcb  = (__bf16*)d_ws;
    __bf16* Qsb  = Qcb + PROJ;
    __bf16* Kcb  = Qcb + 2 * PROJ;
    __bf16* Ksb  = Qcb + 3 * PROJ;
    __bf16* Vb   = Qcb + 4 * PROJ;
    __bf16* kvcT = Qcb + 5 * PROJ;
    __bf16* kvsT = kvcT + KVSZ;
    float* sumcos = (float*)(kvsT + KVSZ);
    float* sumsin = sumcos + NH * HD_DIM;

    dim3 gA(M_TOT / 128, E_DIM / 64);  // 64 x 16
    cosformer_proj_gemm<<<gA, 256, 0, stream>>>(query, Wq, bq, Qcb, Qsb, 1);
    cosformer_proj_gemm<<<gA, 256, 0, stream>>>(key,   Wk, bk, Kcb, Ksb, 1);
    cosformer_proj_gemm<<<gA, 256, 0, stream>>>(value, Wk, bk, Vb,  Vb,  0);

    cosformer_kv_gemm<<<NH, 256, 0, stream>>>(Kcb, Ksb, Vb, kvcT, kvsT);
    cosformer_ksum<<<NH, 256, 0, stream>>>(Kcb, Ksb, sumcos, sumsin);

    dim3 gC(L_SEQ / 64, NH);           // 32 x 64
    cosformer_attn_gemm<<<gC, 256, 0, stream>>>(Qcb, Qsb, kvcT, kvsT,
                                                sumcos, sumsin, out);
}
